// PolymathArgile_40553081209135
// MI455X (gfx1250) — compile-verified, matrix operand order checked by reading, not run on hardware
//
#include <hip/hip_runtime.h>
#include <math.h>

// Problem constants from the reference
#define IN_F   1024
#define OUT_F  1024
#define KNOTS  12
#define EPSF   1e-6f
#define TBL    36      // per-feature table: 12 sorted grid, 12 mc, 12 mt
#define AST    36      // bf16 tile row stride (72B: 8B-aligned chunks, conflict-free banks)

typedef __attribute__((ext_vector_type(16))) __bf16 v16bf;
typedef __attribute__((ext_vector_type(8)))  float  v8f;

struct Chunks32 { uint2 a, b, c, d; };   // 32 bytes == v16bf

// ---- f32 <-> bf16 helpers (explicit RNE, no __bf16 arithmetic needed) ----
static __device__ __forceinline__ unsigned f32_to_bf16_rne(float f) {
  unsigned u = __builtin_bit_cast(unsigned, f);
  unsigned r = 0x7FFFu + ((u >> 16) & 1u);
  return ((u + r) >> 16) & 0xFFFFu;
}
static __device__ __forceinline__ float bf16_to_f32(unsigned h) {
  return __builtin_bit_cast(float, h << 16);
}
// split two f32 into packed bf16 hi-pair and lo-pair (v ~= hi + lo)
static __device__ __forceinline__ void split2(float v0, float v1,
                                              unsigned& hp, unsigned& lp) {
  unsigned h0 = f32_to_bf16_rne(v0);
  unsigned h1 = f32_to_bf16_rne(v1);
  unsigned l0 = f32_to_bf16_rne(v0 - bf16_to_f32(h0));
  unsigned l1 = f32_to_bf16_rne(v1 - bf16_to_f32(h1));
  hp = h0 | (h1 << 16);
  lp = l0 | (l1 << 16);
}

// A/B 16-bit fragment per ISA 7.12.2: lane half -> K chunks [h*8..h*8+7], [16+h*8..]
static __device__ __forceinline__ v16bf load_frag(const unsigned short* rowp, int off) {
  Chunks32 c;
  c.a = *(const uint2*)(rowp + off);
  c.b = *(const uint2*)(rowp + off + 4);
  c.c = *(const uint2*)(rowp + off + 16);
  c.d = *(const uint2*)(rowp + off + 20);
  return __builtin_bit_cast(v16bf, c);
}

// load 4 B fragments (one per 16-wide N sub-tile)
static __device__ __forceinline__ void load_b4(v16bf* dst, const unsigned short* base,
                                               int mlan, int off) {
#pragma unroll
  for (int t = 0; t < 4; ++t)
    dst[t] = load_frag(base + (16 * t + mlan) * AST, off);
}
// 4 WMMAs hitting 4 distinct accumulators (no back-to-back same-dest chains)
static __device__ __forceinline__ void mma4(v8f* acc, v16bf a, const v16bf* b) {
#pragma unroll
  for (int t = 0; t < 4; ++t)
    acc[t] = __builtin_amdgcn_wmma_f32_16x16x32_bf16(
        false, a, false, b[t], (short)0, acc[t], false, false);
}

// ---------------------------------------------------------------------------
// Kernel 1: per-feature knot sort + sigmoid gating -> table in workspace
// ---------------------------------------------------------------------------
__global__ void kan_prep_tables(const float* __restrict__ grid,
                                const float* __restrict__ coeffs,
                                const float* __restrict__ tangents,
                                const float* __restrict__ alive,
                                float* __restrict__ T) {
  int f = blockIdx.x * blockDim.x + threadIdx.x;
  if (f >= IN_F) return;
  float g[KNOTS], c[KNOTS], t[KNOTS], a[KNOTS];
#pragma unroll
  for (int j = 0; j < KNOTS; ++j) {
    g[j] = grid[f * KNOTS + j];
    c[j] = coeffs[f * KNOTS + j];
    t[j] = tangents[f * KNOTS + j];
    a[j] = alive[f * KNOTS + j];
  }
  for (int i = 1; i < KNOTS; ++i) {          // stable insertion sort by grid
    float gk = g[i], ck = c[i], tk = t[i], ak = a[i];
    int j = i - 1;
    while (j >= 0 && g[j] > gk) {
      g[j + 1] = g[j]; c[j + 1] = c[j]; t[j + 1] = t[j]; a[j + 1] = a[j];
      --j;
    }
    g[j + 1] = gk; c[j + 1] = ck; t[j + 1] = tk; a[j + 1] = ak;
  }
#pragma unroll
  for (int j = 0; j < KNOTS; ++j) {
    float s = 1.0f / (1.0f + expf(-a[j]));
    T[f * TBL + j]      = g[j];
    T[f * TBL + 12 + j] = c[j] * s;
    T[f * TBL + 24 + j] = t[j] * s;
  }
}

// ---------------------------------------------------------------------------
// Kernel 2: fused  Y = spline(X) @ P^T + X @ R^T + b
// bf16x3 split-precision GEMM via v_wmma_f32_16x16x32_bf16.
// Block: 256 threads (8 waves). Block tile 128(M) x 64(N); wave tile 16 x 64.
// K-slab = 32 features per iteration.
// ---------------------------------------------------------------------------
__global__ __launch_bounds__(256, 1) void kan_fused_gemm(
    const float* __restrict__ X,   // [M][IN_F]
    const float* __restrict__ T,   // [IN_F][36] spline tables
    const float* __restrict__ P,   // proj_w [OUT_F][IN_F]
    const float* __restrict__ Bv,  // proj_b [OUT_F]
    const float* __restrict__ R,   // res_w  [OUT_F][IN_F]
    float* __restrict__ Y,         // [M][OUT_F]
    int M) {
  __shared__ unsigned short ShS[128 * AST], SlS[128 * AST];  // spline hi/lo
  __shared__ unsigned short XhS[128 * AST], XlS[128 * AST];  // x hi/lo
  __shared__ unsigned short PhS[64 * AST],  PlS[64 * AST];   // proj_w hi/lo
  __shared__ unsigned short RhS[64 * AST],  RlS[64 * AST];   // res_w hi/lo
  __shared__ float Tb[32 * TBL];                             // table slab

  const int tid  = threadIdx.x;
  const int lane = tid & 31;
  const int wave = tid >> 5;           // 0..7 -> M sub-tile
  const int half = (lane >> 4) & 1;    // K-chunk selector (ISA 16-bit layout)
  const int mlan = lane & 15;

  const int rowBase = blockIdx.x * 128;
  const int nBase   = blockIdx.y * 64;

  // staging roles
  const int xr    = tid >> 1;          // 0..127 : x-tile row
  const int cbase = (tid & 1) * 16;    // 16 cols (4 float4s)
  const int wn    = tid >> 2;          // 0..63  : weight-tile row (n)
  const int wc    = (tid & 3) * 8;     // 8 cols (2 float4s)

  int gr = rowBase + xr; if (gr >= M) gr = M - 1;   // safety clamp (load only)

  v8f acc[4];
#pragma unroll
  for (int t = 0; t < 4; ++t)
#pragma unroll
    for (int i = 0; i < 8; ++i) acc[t][i] = 0.0f;

  for (int ko = 0; ko < IN_F; ko += 32) {
    // ---- stage spline-table slab ----
    for (int idx = tid; idx < 32 * TBL; idx += 256)
      Tb[idx] = T[ko * TBL + idx];

    // ---- stage x tile; keep values in regs for spline ----
    const float* xsrc = X + (size_t)gr * IN_F + ko + cbase;
    float xv[16];
    ((float4*)xv)[0] = ((const float4*)xsrc)[0];
    ((float4*)xv)[1] = ((const float4*)xsrc)[1];
    ((float4*)xv)[2] = ((const float4*)xsrc)[2];
    ((float4*)xv)[3] = ((const float4*)xsrc)[3];
#pragma unroll
    for (int j = 0; j < 16; j += 2) {
      unsigned hp, lp;
      split2(xv[j], xv[j + 1], hp, lp);
      *(unsigned*)&XhS[xr * AST + cbase + j] = hp;
      *(unsigned*)&XlS[xr * AST + cbase + j] = lp;
    }

    // ---- stage weight tiles (hi/lo bf16) ----
    const float* psrc = P + (size_t)(nBase + wn) * IN_F + ko + wc;
    const float* rsrc = R + (size_t)(nBase + wn) * IN_F + ko + wc;
    {
      float pv[8], rv[8];
      ((float4*)pv)[0] = ((const float4*)psrc)[0];
      ((float4*)pv)[1] = ((const float4*)psrc)[1];
      ((float4*)rv)[0] = ((const float4*)rsrc)[0];
      ((float4*)rv)[1] = ((const float4*)rsrc)[1];
#pragma unroll
      for (int j = 0; j < 8; j += 2) {
        unsigned hp, lp;
        split2(pv[j], pv[j + 1], hp, lp);
        *(unsigned*)&PhS[wn * AST + wc + j] = hp;
        *(unsigned*)&PlS[wn * AST + wc + j] = lp;
        split2(rv[j], rv[j + 1], hp, lp);
        *(unsigned*)&RhS[wn * AST + wc + j] = hp;
        *(unsigned*)&RlS[wn * AST + wc + j] = lp;
      }
    }

    // prefetch next slab's global sources (global_prefetch_b8)
    if (ko + 32 < IN_F) {
      __builtin_prefetch(xsrc + 32, 0, 3);
      __builtin_prefetch(psrc + 32, 0, 3);
      __builtin_prefetch(rsrc + 32, 0, 3);
    }

    __syncthreads();   // Tb visible

    // ---- spline eval for this thread's 16 x-elements ----
    float sv[16];
#pragma unroll
    for (int cc = 0; cc < 16; ++cc) {
      int c = cbase + cc;
      const float* tb = &Tb[c * TBL];
      float gmin = tb[0], gmax = tb[11];
      float v  = xv[cc];
      float xc = fminf(fmaxf(v, gmin), gmax);
      float gr2 = fmaxf(gmax - gmin, EPSF);
      float xn = (xc - gmin) / gr2 * (float)(KNOTS - 1);
      int il = (int)xn;
      il = il < 0 ? 0 : (il > KNOTS - 2 ? KNOTS - 2 : il);
      float tt = xn - (float)il;
      float p0 = tb[12 + il], p1 = tb[12 + il + 1];
      float m0 = tb[24 + il], m1 = tb[24 + il + 1];
      float xl = tb[il], xrg = tb[il + 1];
      float dx = fmaxf(xrg - xl, EPSF);
      float t2 = tt * tt, t3 = t2 * tt;
      float h00 = 2.0f * t3 - 3.0f * t2 + 1.0f;
      float h10 = t3 - 2.0f * t2 + tt;
      float h01 = 3.0f * t2 - 2.0f * t3;
      float h11 = t3 - t2;
      sv[cc] = h00 * p0 + h01 * p1 + (h10 * m0 + h11 * m1) * dx;
    }
#pragma unroll
    for (int j = 0; j < 16; j += 2) {
      unsigned hp, lp;
      split2(sv[j], sv[j + 1], hp, lp);
      *(unsigned*)&ShS[xr * AST + cbase + j] = hp;
      *(unsigned*)&SlS[xr * AST + cbase + j] = lp;
    }

    __syncthreads();   // all tiles visible

    // ---- bf16x3 MMA: term-outer / tile-inner, double-buffered B frags ----
    const int arow = wave * 16 + mlan;
    const int off  = half * 8;
    v16bf aSh = load_frag(&ShS[arow * AST], off);
    v16bf aSl = load_frag(&SlS[arow * AST], off);
    v16bf aXh = load_frag(&XhS[arow * AST], off);
    v16bf aXl = load_frag(&XlS[arow * AST], off);

    v16bf b0[4], b1[4];
    load_b4(b0, PhS, mlan, off);       // P_hi
    load_b4(b1, PlS, mlan, off);       // P_lo (in flight ahead of use)
    mma4(acc, aSh, b0);                // S_hi * P_hi
    mma4(acc, aSl, b0);                // S_lo * P_hi
    load_b4(b0, RhS, mlan, off);       // R_hi
    mma4(acc, aSh, b1);                // S_hi * P_lo
    load_b4(b1, RlS, mlan, off);       // R_lo
    mma4(acc, aXh, b0);                // X_hi * R_hi
    mma4(acc, aXl, b0);                // X_lo * R_hi
    mma4(acc, aXh, b1);                // X_hi * R_lo

    __syncthreads();   // protect LDS tiles before next slab
  }

  // ---- epilogue: C/D layout (8 VGPRs, lane half -> M+8) -> global, add bias ----
#pragma unroll
  for (int t = 0; t < 4; ++t) {
    const int col = nBase + 16 * t + mlan;
    const float bias = Bv[col];
#pragma unroll
    for (int v = 0; v < 8; ++v) {
      const int row = rowBase + wave * 16 + v + half * 8;
      if (row < M)
        Y[(size_t)row * OUT_F + col] = acc[t][v] + bias;
    }
  }
}

// ---------------------------------------------------------------------------
extern "C" void kernel_launch(void* const* d_in, const int* in_sizes, int n_in,
                              void* d_out, int out_size, void* d_ws, size_t ws_size,
                              hipStream_t stream) {
  const float* x        = (const float*)d_in[0];
  const float* grid     = (const float*)d_in[1];
  const float* coeffs   = (const float*)d_in[2];
  const float* tangents = (const float*)d_in[3];
  const float* alive    = (const float*)d_in[4];
  const float* proj_w   = (const float*)d_in[5];
  const float* proj_b   = (const float*)d_in[6];
  const float* res_w    = (const float*)d_in[7];
  float* out            = (float*)d_out;
  float* tables         = (float*)d_ws;          // 1024*36 floats = 144 KB

  const int M = in_sizes[0] / IN_F;              // 16384

  kan_prep_tables<<<dim3((IN_F + 255) / 256), dim3(256), 0, stream>>>(
      grid, coeffs, tangents, alive, tables);

  dim3 g((M + 127) / 128, OUT_F / 64);
  kan_fused_gemm<<<g, dim3(256), 0, stream>>>(
      x, tables, proj_w, proj_b, res_w, out, M);
}
